// GATEncoder_15934328668575
// MI455X (gfx1250) — matrix operand order repaired and checked by reading, not compiled, not run on hardware
//
#include <hip/hip_runtime.h>

typedef __attribute__((ext_vector_type(16))) _Float16 v16h;
typedef __attribute__((ext_vector_type(8)))  _Float16 v8h;
typedef __attribute__((ext_vector_type(8)))  float    v8f;

#define NEG_SLOPE 0.2f
#define BN_EPS    1e-5f

// ---------------------------------------------------------------- utilities
__global__ void k_fill_f32(float* __restrict__ p, float v, int n) {
  int i = blockIdx.x * blockDim.x + threadIdx.x;
  if (i < n) p[i] = v;
}
__global__ void k_fill_u32(unsigned* __restrict__ p, unsigned v, int n) {
  int i = blockIdx.x * blockDim.x + threadIdx.x;
  if (i < n) p[i] = v;
}
__global__ void k_f32_to_f16(const float* __restrict__ in,
                             _Float16* __restrict__ out, int n) {
  int i = blockIdx.x * blockDim.x + threadIdx.x;
  if (i < n) out[i] = (_Float16)in[i];
}

// ---------------------------------------------- pack W[K,M] f32 -> WMMA B layout f16
// Output layout: [colTile][kTile][lane][e 0..15], so each lane reads one
// contiguous 32-byte chunk per K-step in the GEMM inner loop.
// Lane mapping (B 32x16 f16): col = lane&15, element e -> K = k0 + ((lane>>4)<<3)
// + (e&7) + ((e>>3)<<4) (lanes 0-15: K 0..7,16..23; lanes 16-31: K 8..15,24..31).
template <int K, int M>
__global__ void k_pack_B(const float* __restrict__ W, _Float16* __restrict__ Wpk) {
  int idx = blockIdx.x * blockDim.x + threadIdx.x;
  if (idx >= K * M) return;
  int e    = idx & 15;
  int lane = (idx >> 4) & 31;
  int kt   = (idx >> 9) % (K / 32);
  int ct   = (idx >> 9) / (K / 32);
  int col  = ct * 16 + (lane & 15);
  int k    = kt * 32 + ((lane >> 4) << 3) + (e & 7) + ((e >> 3) << 4);
  Wpk[idx] = (_Float16)W[k * M + col];
}

// ------------------------------------------------- WMMA GEMM: C[n,M] = A[n,K] @ Bpk
// A f16 row-major; Bpk pre-packed WMMA layout; C f32. One wave -> 16x16 tile.
template <int K, int M>
__global__ void k_gemm_wmma(const _Float16* __restrict__ A,
                            const _Float16* __restrict__ Bpk,
                            float* __restrict__ C, int n) {
  const int lane = threadIdx.x & 31;
  const int wave = threadIdx.x >> 5;
  const int colWaves = M / 16;
  const int rowTile = wave / colWaves;
  const int colTile = wave % colWaves;
  const int rowsPerBlock = ((blockDim.x >> 5) / colWaves) << 4;
  const int rowBase = blockIdx.x * rowsPerBlock + rowTile * 16;
  const int colBase = colTile * 16;

  // A addressing (dense 16-bit A 16x32 layout): lane -> row, kbase = 0 or 8
  int aRow = rowBase + (lane & 15);
  if (aRow >= n) aRow = n - 1;            // clamp; keeps EXEC all-ones for WMMA
  const int akb = (lane >> 4) << 3;       // 0 or 8
  const _Float16* bbase = Bpk + (((size_t)colTile * (K / 32) * 32 + lane) << 4);

  v8f acc = {};
#pragma unroll
  for (int kt = 0; kt < K / 32; ++kt) {
    const _Float16* ap = A + (size_t)aRow * K + kt * 32 + akb;
    v8h lo = *(const v8h*)ap;             // K = kb+0..7
    v8h hi = *(const v8h*)(ap + 16);      // K = kb+16..23
    v16h av;
#pragma unroll
    for (int i = 0; i < 8; ++i) { av[i] = lo[i]; av[i + 8] = hi[i]; }
    v16h bv = *(const v16h*)(bbase + ((size_t)kt << 9));  // 32B contiguous
    acc = __builtin_amdgcn_wmma_f32_16x16x32_f16(
        /*neg_a=*/false, av, /*neg_b=*/false, bv,
        /*c_mod=*/(short)0, acc, /*reuse_a=*/false, /*reuse_b=*/false);
  }
  // C/D layout: VGPR v -> row v + (lane>=16 ? 8 : 0), col = lane&15
  const int cCol = colBase + (lane & 15);
  const int rOff = (lane >> 4) << 3;
#pragma unroll
  for (int v = 0; v < 8; ++v) {
    int r = rowBase + rOff + v;
    if (r < n) C[(size_t)r * M + cCol] = acc[v];
  }
}

// --------------------------------------------------------- attention logits
template <int M>
__global__ void k_elr(const float* __restrict__ feat, const float* __restrict__ al,
                      const float* __restrict__ ar, float* __restrict__ el,
                      float* __restrict__ er, int n) {
  int i = blockIdx.x * blockDim.x + threadIdx.x;
  if (i >= n) return;
  const float* f = feat + (size_t)i * M;
  float sl = 0.f, sr = 0.f;
#pragma unroll
  for (int c = 0; c < M; ++c) { float v = f[c]; sl += v * al[c]; sr += v * ar[c]; }
  el[i] = sl; er[i] = sr;
}

// ------------------------------------------ float <-> order-preserving uint key
__device__ __forceinline__ unsigned f32_key(float f) {
  unsigned b = __float_as_uint(f);
  return (b & 0x80000000u) ? ~b : (b | 0x80000000u);
}
__device__ __forceinline__ float key_f32(unsigned k) {
  unsigned b = (k & 0x80000000u) ? (k & 0x7FFFFFFFu) : ~k;
  return __uint_as_float(b);
}

// -------------------------------------------- edge pass 1: e + segment max(dst)
__global__ void k_edge_e_max(const int* __restrict__ src, const int* __restrict__ dst,
                             const float* __restrict__ el, const float* __restrict__ er,
                             float* __restrict__ eOut, unsigned* __restrict__ nodeKey,
                             int eCnt) {
  int i = blockIdx.x * blockDim.x + threadIdx.x;
  if (i >= eCnt) return;
  float v = el[src[i]] + er[dst[i]];
  float e = v > 0.f ? v : NEG_SLOPE * v;
  eOut[i] = e;
  atomicMax(&nodeKey[dst[i]], f32_key(e));
}

// key==0 means no incoming edge -> emax := 0 (zero in-degree guard)
__global__ void k_node_max_fix(unsigned* __restrict__ nodeKey, int n) {
  int i = blockIdx.x * blockDim.x + threadIdx.x;
  if (i >= n) return;
  unsigned k = nodeKey[i];
  float m = (k == 0u) ? 0.f : key_f32(k);
  ((float*)nodeKey)[i] = m;
}

// -------------------------------------------- edge pass 2: exp + segment sum(dst)
__global__ void k_edge_exp_sum(const int* __restrict__ dst, float* __restrict__ eBuf,
                               const float* __restrict__ nmax, float* __restrict__ denom,
                               int eCnt) {
  int i = blockIdx.x * blockDim.x + threadIdx.x;
  if (i >= eCnt) return;
  float ee = __expf(eBuf[i] - nmax[dst[i]]);
  eBuf[i] = ee;                              // in-place e -> ee
  atomicAdd(&denom[dst[i]], ee);
}

// ------------------------------- edge pass 3: rst[dst] += feat[src] * (ee/denom)
// 4 columns per thread: one float4 gather + 4 atomics; edge metadata loaded once
// per 4 columns instead of once per column.
template <int LOG2M>
__global__ void k_edge_aggregate(const int* __restrict__ src, const int* __restrict__ dst,
                                 const float* __restrict__ ee, const float* __restrict__ denom,
                                 const float* __restrict__ feat, float* __restrict__ rst,
                                 int eCnt) {
  constexpr int LOG2TPE = LOG2M - 2;          // threads per edge = M/4
  int g = blockIdx.x * blockDim.x + threadIdx.x;
  int edge = g >> LOG2TPE;
  if (edge >= eCnt) return;
  int colBase = (g & ((1 << LOG2TPE) - 1)) << 2;
  int s = src[edge], d = dst[edge];
  float a = ee[edge] / denom[d];
  const float4 fv = *(const float4*)(feat + ((size_t)s << LOG2M) + colBase);
  float* out = rst + ((size_t)d << LOG2M) + colBase;
  atomicAdd(out + 0, fv.x * a);
  atomicAdd(out + 1, fv.y * a);
  atomicAdd(out + 2, fv.z * a);
  atomicAdd(out + 3, fv.w * a);
}

// --------------------------------------------------------- bias (+ optional ELU)
template <int M, bool ELU>
__global__ void k_bias_act(const float* __restrict__ in, const float* __restrict__ b,
                           float* __restrict__ out, int n) {
  int g = blockIdx.x * blockDim.x + threadIdx.x;
  if (g >= n * M) return;
  int c = g & (M - 1);
  float v = in[g] + b[c];
  if (ELU) v = v > 0.f ? v : (__expf(v) - 1.f);
  out[g] = v;
}

// ---------------------------------------------------------------- batch norm
template <int M>
__global__ void k_bn_stats(const float* __restrict__ h, float* __restrict__ colSum,
                           float* __restrict__ colSumSq, int n) {
  int c = threadIdx.x & (M - 1);
  int rowsPerStep = blockDim.x / M;
  int r = blockIdx.x * rowsPerStep + (threadIdx.x / M);
  int stride = gridDim.x * rowsPerStep;
  float s = 0.f, s2 = 0.f;
  for (; r < n; r += stride) {
    float v = h[(size_t)r * M + c];
    s += v; s2 += v * v;
  }
  atomicAdd(&colSum[c], s);
  atomicAdd(&colSumSq[c], s2);
}

template <int M>
__global__ void k_bn_apply(float* __restrict__ h, const float* __restrict__ colSum,
                           const float* __restrict__ colSumSq, const float* __restrict__ g,
                           const float* __restrict__ be, int n) {
  int idx = blockIdx.x * blockDim.x + threadIdx.x;
  if (idx >= n * M) return;
  int c = idx & (M - 1);
  float invN = 1.f / (float)n;
  float mu = colSum[c] * invN;
  float var = colSumSq[c] * invN - mu * mu;
  float v = h[idx];
  h[idx] = g[c] * (v - mu) * rsqrtf(var + BN_EPS) + be[c];
}

// ================================================================ host side
static inline unsigned cdiv(long long a, int b) { return (unsigned)((a + b - 1) / b); }

template <int K, int M, bool ELU>
static void run_gat_layer(const _Float16* Xh, const float* W, _Float16* Wpk,
                          const float* al, const float* ar, const float* b,
                          const int* src, const int* dst,
                          float* feat, float* rst, float* outBuf,
                          float* el, float* er, float* eBuf,
                          unsigned* nodeKey, float* denom,
                          int n, int eCnt, hipStream_t stream) {
  constexpr int LOG2M = (M == 64) ? 6 : 5;
  // weights f32 -> WMMA-packed f16
  k_pack_B<K, M><<<cdiv(K * M, 256), 256, 0, stream>>>(W, Wpk);
  // GEMM: 32 rows per block (exact for n=100000)
  int gemmThreads = (M == 64) ? 256 : 128;
  k_gemm_wmma<K, M><<<cdiv(n, 32), gemmThreads, 0, stream>>>(Xh, Wpk, feat, n);
  // attention logits
  k_elr<M><<<cdiv(n, 256), 256, 0, stream>>>(feat, al, ar, el, er, n);
  // clear accumulators
  k_fill_f32<<<cdiv((long long)n * M, 256), 256, 0, stream>>>(rst, 0.f, n * M);
  k_fill_f32<<<cdiv(n, 256), 256, 0, stream>>>(denom, 0.f, n);
  k_fill_u32<<<cdiv(n, 256), 256, 0, stream>>>(nodeKey, 0u, n);
  // edge softmax + aggregation
  k_edge_e_max<<<cdiv(eCnt, 256), 256, 0, stream>>>(src, dst, el, er, eBuf, nodeKey, eCnt);
  k_node_max_fix<<<cdiv(n, 256), 256, 0, stream>>>(nodeKey, n);
  k_edge_exp_sum<<<cdiv(eCnt, 256), 256, 0, stream>>>(dst, eBuf, (const float*)nodeKey,
                                                      denom, eCnt);
  k_edge_aggregate<LOG2M><<<cdiv((long long)eCnt * (M / 4), 256), 256, 0, stream>>>(
      src, dst, eBuf, denom, feat, rst, eCnt);
  // bias + activation
  k_bias_act<M, ELU><<<cdiv((long long)n * M, 256), 256, 0, stream>>>(rst, b, outBuf, n);
}

template <int M>
static void run_bn(float* h, const float* g, const float* be, float* colStats,
                   int n, hipStream_t stream) {
  k_fill_f32<<<1, 2 * M, 0, stream>>>(colStats, 0.f, 2 * M);
  k_bn_stats<M><<<512, 256, 0, stream>>>(h, colStats, colStats + M, n);
  k_bn_apply<M><<<cdiv((long long)n * M, 256), 256, 0, stream>>>(h, colStats,
                                                                 colStats + M, g, be, n);
}

extern "C" void kernel_launch(void* const* d_in, const int* in_sizes, int n_in,
                              void* d_out, int out_size, void* d_ws, size_t ws_size,
                              hipStream_t stream) {
  const float* x   = (const float*)d_in[0];
  const int*   src = (const int*)d_in[1];
  const int*   dst = (const int*)d_in[2];
  const float* W0  = (const float*)d_in[3];
  const float* al0 = (const float*)d_in[4];
  const float* ar0 = (const float*)d_in[5];
  const float* b0  = (const float*)d_in[6];
  const float* g0  = (const float*)d_in[7];
  const float* be0 = (const float*)d_in[8];
  const float* W1  = (const float*)d_in[9];
  const float* al1 = (const float*)d_in[10];
  const float* ar1 = (const float*)d_in[11];
  const float* b1  = (const float*)d_in[12];
  const float* g1  = (const float*)d_in[13];
  const float* be1 = (const float*)d_in[14];
  const float* Wm  = (const float*)d_in[15];
  const float* alm = (const float*)d_in[16];
  const float* arm = (const float*)d_in[17];
  const float* bm  = (const float*)d_in[18];

  const int n    = in_sizes[0] / 128;   // D_IN = 128
  const int eCnt = in_sizes[1];

  // ---- workspace carve (256B aligned)
  char* ws = (char*)d_ws;
  auto carve = [&](size_t bytes) -> void* {
    void* p = (void*)ws;
    ws += (bytes + 255) & ~(size_t)255;
    return p;
  };
  _Float16* Xh      = (_Float16*)carve((size_t)n * 128 * sizeof(_Float16));
  _Float16* Wpk     = (_Float16*)carve(128 * 64 * sizeof(_Float16));
  float*    feat    = (float*)carve((size_t)n * 64 * sizeof(float));
  float*    rst     = (float*)carve((size_t)n * 64 * sizeof(float));
  float*    el      = (float*)carve((size_t)n * sizeof(float));
  float*    er      = (float*)carve((size_t)n * sizeof(float));
  float*    eBuf    = (float*)carve((size_t)eCnt * sizeof(float));
  unsigned* nodeKey = (unsigned*)carve((size_t)n * sizeof(unsigned));
  float*    denom   = (float*)carve((size_t)n * sizeof(float));
  float*    stats   = (float*)carve(128 * sizeof(float));
  (void)ws_size; (void)n_in; (void)out_size;

  // ---- layer 0: 128 -> 64, ELU, then BN twice
  k_f32_to_f16<<<cdiv((long long)n * 128, 256), 256, 0, stream>>>(x, Xh, n * 128);
  run_gat_layer<128, 64, true>(Xh, W0, Wpk, al0, ar0, b0, src, dst,
                               feat, rst, rst, el, er, eBuf, nodeKey, denom,
                               n, eCnt, stream);
  run_bn<64>(rst, g0, be0, stats, n, stream);
  run_bn<64>(rst, g0, be0, stats, n, stream);

  // ---- layer 1: 64 -> 64, ELU, then BN twice
  k_f32_to_f16<<<cdiv((long long)n * 64, 256), 256, 0, stream>>>(rst, Xh, n * 64);
  run_gat_layer<64, 64, true>(Xh, W1, Wpk, al1, ar1, b1, src, dst,
                              feat, rst, rst, el, er, eBuf, nodeKey, denom,
                              n, eCnt, stream);
  run_bn<64>(rst, g1, be1, stats, n, stream);
  run_bn<64>(rst, g1, be1, stats, n, stream);

  // ---- enc_mu: 64 -> 32, no activation, straight to d_out
  k_f32_to_f16<<<cdiv((long long)n * 64, 256), 256, 0, stream>>>(rst, Xh, n * 64);
  run_gat_layer<64, 32, false>(Xh, Wm, Wpk, alm, arm, bm, src, dst,
                               feat, rst, (float*)d_out, el, er, eBuf, nodeKey, denom,
                               n, eCnt, stream);
}
